// MultiHeadAttention_16209206575111
// MI455X (gfx1250) — matrix operand order repaired and checked by reading, not compiled, not run on hardware
//
#include <hip/hip_runtime.h>

// ---------------------------------------------------------------------------
// MHA forward for B=2, S=2048, D=1024, H=16, DH=64 on gfx1250 (MI455X).
// All GEMMs use v_wmma_f32_16x16x32_bf16 (fp32 accumulate); fp32 -> bf16
// conversion happens during LDS staging. Attention K/V/Q tiles are moved with
// GLOBAL_LOAD_ASYNC_TO_LDS_B128 (ASYNCcnt) when the builtin is available.
// ---------------------------------------------------------------------------

typedef __attribute__((ext_vector_type(16))) __bf16 v16bf;
typedef __attribute__((ext_vector_type(8)))  __bf16 v8bf;
typedef __attribute__((ext_vector_type(8)))  float  v8f;
typedef int i32x4 __attribute__((vector_size(16)));

#define B_  2
#define S_  2048
#define D_  1024
#define H_  16
#define DH_ 64
#define M_  (B_ * S_)   // 4096

#define AS1 __attribute__((address_space(1)))
#define AS3 __attribute__((address_space(3)))

#if __has_builtin(__builtin_amdgcn_global_load_async_to_lds_b128)
#define HAVE_ASYNC_LDS 1
#else
#define HAVE_ASYNC_LDS 0
#endif

// 16-byte global -> LDS copy: async DMA if available, else vector copy.
__device__ __forceinline__ void cp16_g2l(__bf16* lds, const __bf16* g) {
#if HAVE_ASYNC_LDS
    __builtin_amdgcn_global_load_async_to_lds_b128(
        (AS1 i32x4*)const_cast<__bf16*>(g), (AS3 i32x4*)lds, 0, 0);
#else
    *(uint4*)lds = *(const uint4*)g;
#endif
}

__device__ __forceinline__ void async_wait_all() {
#if HAVE_ASYNC_LDS
#if __has_builtin(__builtin_amdgcn_s_wait_asynccnt)
    __builtin_amdgcn_s_wait_asynccnt(0);
#else
    asm volatile("s_wait_asynccnt 0x0" ::: "memory");
#endif
#endif
}

union FragU { v16bf v; v8bf h[2]; };

// Load one 16-value bf16 fragment as two contiguous 16B LDS chunks.
__device__ __forceinline__ v16bf load_frag(const __bf16* p0, const __bf16* p1) {
    FragU u;
    u.h[0] = *(const v8bf*)p0;
    u.h[1] = *(const v8bf*)p1;
    return u.v;
}

__device__ __forceinline__ v8f wmma_bf16(v16bf a, v16bf b, v8f c) {
    return __builtin_amdgcn_wmma_f32_16x16x32_bf16(false, a, false, b, (short)0, c,
                                                   false, false);
}

__device__ __forceinline__ float rowmax16(float v) {
    v = fmaxf(v, __shfl_xor(v, 1, 32));
    v = fmaxf(v, __shfl_xor(v, 2, 32));
    v = fmaxf(v, __shfl_xor(v, 4, 32));
    v = fmaxf(v, __shfl_xor(v, 8, 32));
    return v;
}
__device__ __forceinline__ float rowsum16(float v) {
    v += __shfl_xor(v, 1, 32);
    v += __shfl_xor(v, 2, 32);
    v += __shfl_xor(v, 4, 32);
    v += __shfl_xor(v, 8, 32);
    return v;
}

// ---------------------------------------------------------------------------
// GEMM: Y = X(fp32)[M x K] * W(fp32)[K x N] + bias, via bf16 WMMA.
// MODE 0: bf16 output, head-split [B,H,S,DH]            (Q,K projections)
// MODE 1: fp32 output, flat [M,N]                       (output projection)
// MODE 2: bf16 output, head-split transposed [B,H,DH,S] (V projection)
// Block: 256 threads (8 waves), tile 128x128, k-step 32, double-buffered LDS.
// Wave grid 2(M) x 4(N): each wave owns 64x32 = 4x2 subtiles of 16x16.
// ---------------------------------------------------------------------------
template <int MODE>
__global__ __launch_bounds__(256) void gemm_bf16_k(
    const float* __restrict__ X, const float* __restrict__ W,
    const float* __restrict__ bias, void* __restrict__ out) {
    constexpr int TK = 32;
    __shared__ __attribute__((aligned(16))) __bf16 lA[2][128 * TK];  // [m][k]
    __shared__ __attribute__((aligned(16))) __bf16 lB[2][128 * TK];  // [n][k]

    const int tid   = threadIdx.x;
    const int lane  = tid & 31;
    const int wave  = tid >> 5;
    const int waveM = wave >> 2;  // 0..1
    const int waveN = wave & 3;   // 0..3
    const int hi    = lane >> 4;
    const int l16   = lane & 15;
    const int m0    = blockIdx.y * 128;
    const int n0    = blockIdx.x * 128;

    auto stageAB = [&](int buf, int k0) {
        // A tile 128x32 fp32 -> bf16, packed 8B LDS stores.
#pragma unroll
        for (int j = 0; j < 4; ++j) {
            const int c   = tid + j * 256;  // 0..1023 float4 chunks
            const int row = c >> 3;
            const int kq  = (c & 7) << 2;
            const float4 f = *(const float4*)(X + (size_t)(m0 + row) * D_ + k0 + kq);
            union { __bf16 b[4]; uint2 u; } t;
            t.b[0] = (__bf16)f.x; t.b[1] = (__bf16)f.y;
            t.b[2] = (__bf16)f.z; t.b[3] = (__bf16)f.w;
            *(uint2*)&lA[buf][row * TK + kq] = t.u;
        }
        // B tile transposed into [n][k] (coalesced global reads over n).
#pragma unroll
        for (int j = 0; j < 4; ++j) {
            const int c  = tid + j * 256;
            const int k  = c >> 5;
            const int n4 = (c & 31) << 2;
            const float4 f = *(const float4*)(W + (size_t)(k0 + k) * D_ + n0 + n4);
            lB[buf][(n4 + 0) * TK + k] = (__bf16)f.x;
            lB[buf][(n4 + 1) * TK + k] = (__bf16)f.y;
            lB[buf][(n4 + 2) * TK + k] = (__bf16)f.z;
            lB[buf][(n4 + 3) * TK + k] = (__bf16)f.w;
        }
    };

    v8f acc[4][2];
#pragma unroll
    for (int mi = 0; mi < 4; ++mi)
#pragma unroll
        for (int ni = 0; ni < 2; ++ni)
            acc[mi][ni] = (v8f){0.f, 0.f, 0.f, 0.f, 0.f, 0.f, 0.f, 0.f};

    stageAB(0, 0);
    __syncthreads();

    constexpr int NKT = D_ / TK;  // 32
    for (int kt = 0; kt < NKT; ++kt) {
        const int cur = kt & 1;
        if (kt + 1 < NKT) stageAB(cur ^ 1, (kt + 1) * TK);
        if (kt + 2 < NKT) {
            __builtin_prefetch(X + (size_t)(m0 + (tid >> 1)) * D_ + (kt + 2) * TK, 0, 1);
            __builtin_prefetch(W + (size_t)((kt + 2) * TK + (tid >> 3)) * D_ + n0, 0, 1);
        }

        const __bf16* Ab = lA[cur];
        const __bf16* Bb = lB[cur];
        v16bf aF[4];
#pragma unroll
        for (int mi = 0; mi < 4; ++mi) {
            const int m = waveM * 64 + mi * 16 + l16;
            aF[mi] = load_frag(&Ab[m * TK + hi * 8], &Ab[m * TK + 16 + hi * 8]);
        }
#pragma unroll
        for (int ni = 0; ni < 2; ++ni) {
            const int n = waveN * 32 + ni * 16 + l16;
            const v16bf bF = load_frag(&Bb[n * TK + hi * 8], &Bb[n * TK + 16 + hi * 8]);
#pragma unroll
            for (int mi = 0; mi < 4; ++mi)
                acc[mi][ni] = wmma_bf16(aF[mi], bF, acc[mi][ni]);
        }
        __syncthreads();
    }

    // Epilogue: bias add + store.
#pragma unroll
    for (int ni = 0; ni < 2; ++ni) {
        const int gn = n0 + waveN * 32 + ni * 16 + l16;
        const float bv = bias[gn];
#pragma unroll
        for (int mi = 0; mi < 4; ++mi) {
#pragma unroll
            for (int r = 0; r < 8; ++r) {
                const int gm = m0 + waveM * 64 + mi * 16 + hi * 8 + r;
                const float y = acc[mi][ni][r] + bv;
                const int b = gm >> 11, s = gm & (S_ - 1);
                const int h = gn >> 6, dh = gn & (DH_ - 1);
                if (MODE == 0) {
                    ((__bf16*)out)[(((size_t)(b * H_ + h) * S_ + s) << 6) + dh] =
                        (__bf16)y;
                } else if (MODE == 2) {
                    ((__bf16*)out)[((size_t)(b * H_ + h) * DH_ + dh) * S_ + s] =
                        (__bf16)y;
                } else {
                    ((float*)out)[(size_t)gm * D_ + gn] = y;
                }
            }
        }
    }
}

// ---------------------------------------------------------------------------
// Flash attention over one (b, h, 64-query tile). 128 threads = 4 waves,
// each wave owns 16 query rows. K/V tiles double-buffered via async LDS DMA;
// online softmax; both GEMMs via bf16 WMMA. V arrives pre-transposed
// [B,H,DH,S]. Writes fp32 ctx in [B,S,D] layout.
// ---------------------------------------------------------------------------
__global__ __launch_bounds__(128) void attn_k(
    const __bf16* __restrict__ Q, const __bf16* __restrict__ Kh,
    const __bf16* __restrict__ Vt, const float* __restrict__ mask,
    float* __restrict__ ctx) {
    const int qt = blockIdx.x & 31;          // S/64 = 32 query tiles
    const int h  = (blockIdx.x >> 5) & 15;
    const int b  = blockIdx.x >> 9;

    __shared__ __attribute__((aligned(16))) __bf16 sQ[64 * DH_];       // [m][dh]
    __shared__ __attribute__((aligned(16))) __bf16 sK[2][64 * DH_];    // [key][dh]
    __shared__ __attribute__((aligned(16))) __bf16 sVt[2][DH_ * 64];   // [dh][key]
    __shared__ __attribute__((aligned(16))) __bf16 sP[4 * 16 * 64];

    const int tid  = threadIdx.x;
    const int lane = tid & 31;
    const int wave = tid >> 5;
    const int hi   = lane >> 4;
    const int l16  = lane & 15;

    const size_t headBase = (size_t)(b * H_ + h) * S_ * DH_;  // Q,K and Vt base

    // Issue async copies for K tile (row-major) and Vt tile (dh-major).
    auto issueKV = [&](int kt, int buf) {
#pragma unroll
        for (int j = 0; j < 4; ++j) {
            const int c = tid + j * 128;  // 0..511 chunks of 8 bf16
            cp16_g2l(&sK[buf][c * 8],
                     Kh + headBase + (size_t)kt * 64 * DH_ + c * 8);
            const int dh = c >> 3, part = c & 7;
            cp16_g2l(&sVt[buf][dh * 64 + part * 8],
                     Vt + headBase + (size_t)dh * S_ + kt * 64 + part * 8);
        }
    };

    // Prologue: Q tile + first K/V tile.
#pragma unroll
    for (int j = 0; j < 4; ++j) {
        const int c = tid + j * 128;
        cp16_g2l(&sQ[c * 8], Q + headBase + (size_t)qt * 64 * DH_ + c * 8);
    }
    issueKV(0, 0);
    async_wait_all();
    __syncthreads();

    // Q fragments are invariant over the kv loop.
    v16bf aQ[2];
    {
        const int m = wave * 16 + l16;
#pragma unroll
        for (int ks = 0; ks < 2; ++ks)
            aQ[ks] = load_frag(&sQ[m * DH_ + ks * 32 + hi * 8],
                               &sQ[m * DH_ + ks * 32 + 16 + hi * 8]);
    }

    v8f o[4];
#pragma unroll
    for (int nt = 0; nt < 4; ++nt)
        o[nt] = (v8f){0.f, 0.f, 0.f, 0.f, 0.f, 0.f, 0.f, 0.f};
    float rm[8], rl[8];
#pragma unroll
    for (int r = 0; r < 8; ++r) { rm[r] = -3.0e38f; rl[r] = 0.f; }

    const float scale = 0.125f;  // 1/sqrt(64)
    constexpr int NT = S_ / 64;  // 32 kv tiles

    for (int kt = 0; kt < NT; ++kt) {
        const int cur = kt & 1;
        if (kt + 1 < NT) issueKV(kt + 1, cur ^ 1);  // overlap with compute

        // Scores S = (Q K^T) * scale + mask for this wave's 16x64 slab.
        const __bf16* Kb = sK[cur];
        v8f sc[4];
#pragma unroll
        for (int ni = 0; ni < 4; ++ni) {
            sc[ni] = (v8f){0.f, 0.f, 0.f, 0.f, 0.f, 0.f, 0.f, 0.f};
            const int n = ni * 16 + l16;  // key within tile
#pragma unroll
            for (int ks = 0; ks < 2; ++ks) {
                const v16bf bK = load_frag(&Kb[n * DH_ + ks * 32 + hi * 8],
                                           &Kb[n * DH_ + ks * 32 + 16 + hi * 8]);
                sc[ni] = wmma_bf16(aQ[ks], bK, sc[ni]);
            }
            const float mv = mask[b * S_ + kt * 64 + n] * (-1.0e9f);
#pragma unroll
            for (int r = 0; r < 8; ++r) sc[ni][r] = sc[ni][r] * scale + mv;
        }

        // Online softmax per row (rows split across the two 16-lane halves).
#pragma unroll
        for (int r = 0; r < 8; ++r) {
            float mloc = fmaxf(fmaxf(sc[0][r], sc[1][r]), fmaxf(sc[2][r], sc[3][r]));
            mloc = rowmax16(mloc);
            const float mnew = fmaxf(rm[r], mloc);
            const float ef   = __expf(rm[r] - mnew);
            const float p0 = __expf(sc[0][r] - mnew);
            const float p1 = __expf(sc[1][r] - mnew);
            const float p2 = __expf(sc[2][r] - mnew);
            const float p3 = __expf(sc[3][r] - mnew);
            const float ls = rowsum16(p0 + p1 + p2 + p3);
            rl[r] = rl[r] * ef + ls;
            rm[r] = mnew;
            o[0][r] *= ef; o[1][r] *= ef; o[2][r] *= ef; o[3][r] *= ef;
            // Spill P to LDS (C-layout -> A-layout conversion).
            const int m = r + hi * 8;
            __bf16* pp = &sP[wave * 1024 + m * 64 + l16];
            pp[0]  = (__bf16)p0;
            pp[16] = (__bf16)p1;
            pp[32] = (__bf16)p2;
            pp[48] = (__bf16)p3;
        }
        __syncthreads();  // order sP stores before sP fragment loads

        // O += P * V
        const __bf16* Vb = sVt[cur];
#pragma unroll
        for (int ks = 0; ks < 2; ++ks) {
            const v16bf aP =
                load_frag(&sP[wave * 1024 + l16 * 64 + ks * 32 + hi * 8],
                          &sP[wave * 1024 + l16 * 64 + ks * 32 + 16 + hi * 8]);
#pragma unroll
            for (int nt = 0; nt < 4; ++nt) {
                const int n = nt * 16 + l16;  // dh
                const v16bf bV = load_frag(&Vb[n * 64 + ks * 32 + hi * 8],
                                           &Vb[n * 64 + ks * 32 + 16 + hi * 8]);
                o[nt] = wmma_bf16(aP, bV, o[nt]);
            }
        }
        async_wait_all();  // next tile's DMA complete
        __syncthreads();   // all waves done reading current buffers
    }

    // Normalize and write ctx (fp32, [B,S,D], heads re-interleaved).
#pragma unroll
    for (int nt = 0; nt < 4; ++nt) {
        const int col = h * DH_ + nt * 16 + l16;
#pragma unroll
        for (int r = 0; r < 8; ++r) {
            const int gm = b * S_ + qt * 64 + wave * 16 + hi * 8 + r;
            ctx[(size_t)gm * D_ + col] = o[nt][r] / rl[r];
        }
    }
}

// ---------------------------------------------------------------------------
extern "C" void kernel_launch(void* const* d_in, const int* in_sizes, int n_in,
                              void* d_out, int out_size, void* d_ws, size_t ws_size,
                              hipStream_t stream) {
    const float* query = (const float*)d_in[0];
    const float* key   = (const float*)d_in[1];
    const float* value = (const float*)d_in[2];
    const float* mask  = (const float*)d_in[3];
    const float* Wq    = (const float*)d_in[4];
    const float* bq    = (const float*)d_in[5];
    const float* Wk    = (const float*)d_in[6];
    const float* bk    = (const float*)d_in[7];
    const float* Wv    = (const float*)d_in[8];
    const float* bv    = (const float*)d_in[9];
    const float* Wo    = (const float*)d_in[10];
    const float* bo    = (const float*)d_in[11];

    const size_t nElem = (size_t)M_ * D_;  // 4M
    __bf16* Qh  = (__bf16*)d_ws;
    __bf16* Kh  = Qh + nElem;
    __bf16* Vth = Kh + nElem;              // [B,H,DH,S]
    float*  ctx = (float*)(Vth + nElem);

    dim3 gGemm(D_ / 128, M_ / 128);  // (8, 32)
    dim3 bGemm(256);

    gemm_bf16_k<0><<<gGemm, bGemm, 0, stream>>>(query, Wq, bq, Qh);
    gemm_bf16_k<0><<<gGemm, bGemm, 0, stream>>>(key,   Wk, bk, Kh);
    gemm_bf16_k<2><<<gGemm, bGemm, 0, stream>>>(value, Wv, bv, Vth);

    attn_k<<<dim3(B_ * H_ * (S_ / 64)), dim3(128), 0, stream>>>(Qh, Kh, Vth, mask, ctx);

    gemm_bf16_k<1><<<gGemm, bGemm, 0, stream>>>(ctx, Wo, bo, d_out);
}
